// Convolution_1176821039249
// MI455X (gfx1250) — matrix operand order repaired and checked by reading, not compiled, not run on hardware
//
#include <hip/hip_runtime.h>

// ---------------------------------------------------------------------------
// Multi-scale box filter (k = 3,5,7,9,11,13,15), replicate padding, averaged,
// then multiplied by base_map.  4096x4096 fp32.
//
// MI455X plan:
//  * memory-bound (~192 MB min traffic -> ~8.2 us @ 23.3 TB/s): input touches
//    HBM exactly once; halo re-reads hit the 192 MB L2.
//  * interior tiles: Tensor Data Mover (tensor_load_to_lds) DMAs the 46x78
//    fp32 tile into LDS in ONE instruction (TENSORcnt-tracked), issued by one
//    wave and overlapped with the constant A-matrix build. Border tiles use
//    clamped per-thread loads (replicate pad).
//  * horizontal windowed sums for all 7 scales via sliding accumulation.
//  * vertical combine for ALL scales via WMMA: Out16x16 = sum_k V_k * H_k,
//    V_k = constant 16x30 banded matrix (1/k^2 in band), one
//    v_wmma_f32_16x16x32_f16 per scale per chunk, f32 accumulation.
// ---------------------------------------------------------------------------

typedef _Float16 v16h __attribute__((ext_vector_type(16)));
typedef _Float16 v8h  __attribute__((ext_vector_type(8)));
typedef float    v8f  __attribute__((ext_vector_type(8)));
typedef unsigned int u32x4 __attribute__((ext_vector_type(4)));
typedef int      i32x4 __attribute__((ext_vector_type(4)));
typedef int      i32x8 __attribute__((ext_vector_type(8)));

#define IMG_H 4096
#define IMG_W 4096
#define PAD   7

#define TOUT_R 32
#define TOUT_C 64
#define TIN_R  (TOUT_R + 2 * PAD)   // 46
#define TIN_C  (TOUT_C + 2 * PAD)   // 78  (contiguous rows: TDM-friendly)
#define HF_R   48                   // rows 46,47 zeroed (K padding for wmma)
#define HF_RP  56                   // f16 row pitch: 112B lane stride, 16B aligned

constexpr int   KS[7] = {3, 5, 7, 9, 11, 13, 15};
constexpr float KW[7] = {1.0f/9.0f,  1.0f/25.0f,  1.0f/49.0f, 1.0f/81.0f,
                         1.0f/121.0f, 1.0f/169.0f, 1.0f/225.0f};

#if __has_include(<hip/amd_detail/amd_gfx1250_TDM.h>)
  #define TDM_ARITY6 1   // therock-10.0 headers -> 6-arg builtin
#endif

static __device__ __forceinline__ void tdm_wait_zero() {
#if __has_builtin(__builtin_amdgcn_s_wait_tensorcnt)
    __builtin_amdgcn_s_wait_tensorcnt(0);
#else
    asm volatile("s_wait_tensorcnt 0x0" ::: "memory");
#endif
}

// One-shot DMA of a 46x78 fp32 tile (row stride IMG_W) from global into LDS.
// D# built per CDNA5 ISA 8.3/8.4: group0 = {flags, lds_addr, global_addr,
// type=2}; group1 = {mask=0, data_size=4B, dims/strides, tile dims}.
// NOTE: called unconditionally (no __has_builtin guard) so a missing builtin
// or an arity mismatch produces a precise compile diagnostic instead of
// silently falling back.
static __device__ __forceinline__ void tdm_load_tile(const float* gsrc,
                                                     unsigned lds_byte_off) {
    unsigned long long ga = (unsigned long long)(uintptr_t)gsrc;

    u32x4 g0;
    g0[0] = 1u;                                        // count=1, user mode
    g0[1] = lds_byte_off;                              // lds_addr (bytes)
    g0[2] = (unsigned)ga;                              // global_addr[31:0]
    g0[3] = (unsigned)((ga >> 32) & 0x1FFFFFFull)      // global_addr[56:32]
          | (2u << 30);                                // type = 2 ("image")

    // group1 bitfields packed via four 64-bit words (bits 0..255)
    unsigned long long q0 = (2ull << 16)                               // data_size=4B
                          | ((unsigned long long)(IMG_W & 0xFFFF) << 48); // tensor_dim0 lo
    unsigned long long q1 = ((unsigned long long)(IMG_W >> 16))        // tensor_dim0 hi
                          | ((unsigned long long)IMG_H << 16)          // tensor_dim1
                          | ((unsigned long long)TIN_C << 48);         // tile_dim0 = 78
    unsigned long long q2 = (unsigned long long)TIN_R                  // tile_dim1 = 46
                          | ((unsigned long long)IMG_W << 32);         // dim0_stride lo
    unsigned long long q3 = ((unsigned long long)IMG_W * IMG_H) << 16; // dim1_stride

    i32x8 g1;
    g1[0] = (int)q0; g1[1] = (int)(q0 >> 32);
    g1[2] = (int)q1; g1[3] = (int)(q1 >> 32);
    g1[4] = (int)q2; g1[5] = (int)(q2 >> 32);
    g1[6] = (int)q3; g1[7] = (int)(q3 >> 32);

    i32x4 z4 = {0, 0, 0, 0};
#if defined(TDM_ARITY6)
    i32x8 z8 = {0, 0, 0, 0, 0, 0, 0, 0};
    __builtin_amdgcn_tensor_load_to_lds(g0, g1, z4, z4, z8, 0);
#else
    __builtin_amdgcn_tensor_load_to_lds(g0, g1, z4, z4, 0);
#endif
}

__global__ __launch_bounds__(256) void
multibox_wmma_kernel(const float* __restrict__ x,
                     const float* __restrict__ bmap,
                     float* __restrict__ out)
{
    __shared__ float    s_in[TIN_R][TIN_C];
    __shared__ _Float16 s_h[7][TOUT_C][HF_RP];   // [scale][col][row], col-major

    const int tid  = threadIdx.x;
    const int row0 = blockIdx.y * TOUT_R;
    const int col0 = blockIdx.x * TOUT_C;
    const int lane = tid & 31;
    const int wave = tid >> 5;

    // hint the base_map tile into cache early (global_prefetch_b8)
    __builtin_prefetch(&bmap[(size_t)(row0 + (tid >> 3)) * IMG_W + col0 + ((tid & 7) << 3)], 0, 1);

    // ---- stage 0: tile load (TDM for interior tiles, clamped loads else) ---
    const bool interior =
        (blockIdx.x > 0) && (blockIdx.x < (IMG_W / TOUT_C) - 1) &&
        (blockIdx.y > 0) && (blockIdx.y < (IMG_H / TOUT_R) - 1);
    if (interior) {
        if (wave == 0) {
            const float* gsrc = x + (size_t)(row0 - PAD) * IMG_W + (col0 - PAD);
            tdm_load_tile(gsrc, (unsigned)(uintptr_t)&s_in[0][0]);
        }
    } else {
        for (int idx = tid; idx < TIN_R * TIN_C; idx += 256) {
            int r = idx / TIN_C;
            int c = idx - r * TIN_C;
            int gr = row0 - PAD + r;
            int gc = col0 - PAD + c;
            gr = gr < 0 ? 0 : (gr > IMG_H - 1 ? IMG_H - 1 : gr);
            gc = gc < 0 ? 0 : (gc > IMG_W - 1 ? IMG_W - 1 : gc);
            s_in[r][c] = x[(size_t)gr * IMG_W + gc];
        }
    }

    // ---- overlap with DMA: build constant banded A matrices (registers) ----
    // 16-bit A layout (16x32): lane L holds M = L&15; VGPR v holds halves
    // (K, K+1) with K = 2v (+8 for v>=4) + 8*(L>>4).
    const int m    = lane & 15;
    const int koff = (lane >> 4) << 3;
    v16h A[7];
    #pragma unroll
    for (int kidx = 0; kidx < 7; ++kidx) {
        const int k  = KS[kidx];
        const int h  = k >> 1;
        const _Float16 w = (_Float16)KW[kidx];
        const int lo = m + PAD - h;          // band: lo <= K <= lo + k - 1
        #pragma unroll
        for (int v = 0; v < 8; ++v) {
            int kb = 2 * v + (v >= 4 ? 8 : 0) + koff;
            A[kidx][2 * v]     = ((unsigned)(kb     - lo) < (unsigned)k) ? w : (_Float16)0.0f;
            A[kidx][2 * v + 1] = ((unsigned)(kb + 1 - lo) < (unsigned)k) ? w : (_Float16)0.0f;
        }
    }

    if (interior && wave == 0) tdm_wait_zero();   // TENSORcnt -> 0
    __syncthreads();

    // ---- stage 1: horizontal windowed sums, all 7 scales (sliding) ---------
    for (int p = tid; p < 7 * HF_R; p += 256) {
        int kidx = p / HF_R;
        int r    = p - kidx * HF_R;
        if (r >= TIN_R) {                 // zero the K-padding rows 46,47
            for (int c = 0; c < TOUT_C; ++c)
                s_h[kidx][c][r] = (_Float16)0.0f;
            continue;
        }
        int k = KS[kidx];
        int h = k >> 1;
        float s = 0.0f;
        #pragma unroll 1
        for (int j = PAD - h; j < PAD + h; ++j)
            s += s_in[r][j];
        #pragma unroll 1
        for (int c = 0; c < TOUT_C; ++c) {
            s += s_in[r][c + PAD + h];    // window [c+PAD-h, c+PAD+h], len k
            s_h[kidx][c][r] = (_Float16)s;
            s -= s_in[r][c + PAD - h];
        }
    }
    __syncthreads();

    // ---- stage 2: vertical multi-scale combine via WMMA --------------------
    // 8 waves -> 8 chunks of 16x16 outputs (2 row-chunks x 4 col-chunks)
    const int R  = (wave >> 2) << 4;
    const int c0 = (wave & 3) << 4;

    // B (32x16 f16): lane L holds N = L&15, K = 8*(L>>4) + 0..7 and
    // 16 + 8*(L>>4) + 0..7 at fixed column in the col-major h-field.
    const int n     = lane & 15;
    const int rbase = R + ((lane >> 4) << 3);

    v8f acc = {};
    #pragma unroll
    for (int kidx = 0; kidx < 7; ++kidx) {
        const _Float16* bp = &s_h[kidx][c0 + n][rbase];
        v8h blo = *(const v8h*)(bp);
        v8h bhi = *(const v8h*)(bp + 16);
        v16h b;
        #pragma unroll
        for (int i = 0; i < 8; ++i) { b[i] = blo[i]; b[i + 8] = bhi[i]; }
        acc = __builtin_amdgcn_wmma_f32_16x16x32_f16(
                  false, A[kidx], false, b, (short)0, acc, false, false);
    }

    // ---- stage 3: scale by 1/7, multiply base_map, store -------------------
    const int mrow = ((lane >> 4) << 3);   // C/D: VGPR v -> M = v + 8*(L>>4)
    #pragma unroll
    for (int v = 0; v < 8; ++v) {
        int gr = row0 + R + mrow + v;
        int gc = col0 + c0 + n;
        size_t o = (size_t)gr * IMG_W + gc;
        out[o] = acc[v] * (1.0f / 7.0f) * bmap[o];
    }
}

extern "C" void kernel_launch(void* const* d_in, const int* in_sizes, int n_in,
                              void* d_out, int out_size, void* d_ws, size_t ws_size,
                              hipStream_t stream)
{
    (void)in_sizes; (void)n_in; (void)out_size; (void)d_ws; (void)ws_size;
    const float* x    = (const float*)d_in[0];
    const float* bmap = (const float*)d_in[1];
    float* out        = (float*)d_out;

    dim3 grid(IMG_W / TOUT_C, IMG_H / TOUT_R);   // 64 x 128 blocks
    dim3 block(256);
    multibox_wmma_kernel<<<grid, block, 0, stream>>>(x, bmap, out);
}
